// WSSSAffinityLoss_81192061764317
// MI455X (gfx1250) — matrix operand order repaired and checked by reading, not compiled, not run on hardware
//
#include <hip/hip_runtime.h>

typedef __attribute__((ext_vector_type(16))) _Float16     v16h;
typedef __attribute__((ext_vector_type(8)))  float        v8f;
typedef __attribute__((ext_vector_type(8)))  unsigned int v8u;

#define B_   2
#define K_   8
#define C_   256
#define HW_  16384
#define S_   2048
#define TILE 16
#define SIM_TH 0.95f

// workspace layout
#define FH_BYTES ((size_t)B_ * S_ * C_ * 2)   // f16 normalized features: 2 MiB
#define MS_BYTES ((size_t)B_ * K_ * S_ * 4)   // gathered sigmoid probs: 128 KiB

__global__ void zero_accum_kernel(float* acc) {
  if (threadIdx.x < 2) acc[threadIdx.x] = 0.f;
}

// One wave (32 lanes) per (b, sample): gather a C=256 column, L2-normalize,
// store as a contiguous f16 row Fh[b][j][0..255].
__global__ void __launch_bounds__(32)
gather_normalize_kernel(const float* __restrict__ feats,
                        const int* __restrict__ indices,
                        _Float16* __restrict__ Fh) {
  int wid  = blockIdx.x;            // 0 .. B*S-1
  int b    = wid / S_;
  int j    = wid % S_;
  int lane = threadIdx.x;
  int p    = indices[b * S_ + j];
  const float* fb = feats + (size_t)b * C_ * HW_ + p;

  float v[8];
  float ss = 0.f;
#pragma unroll
  for (int u = 0; u < 8; ++u) {
    int c = lane * 8 + u;
    float x = fb[(size_t)c * HW_];
    v[u] = x;
    ss += x * x;
  }
#pragma unroll
  for (int off = 16; off >= 1; off >>= 1) ss += __shfl_xor(ss, off, 32);
  float scale = 1.f / fmaxf(sqrtf(ss), 1e-12f);

  union { _Float16 h[8]; uint4 q; } o;
#pragma unroll
  for (int u = 0; u < 8; ++u) o.h[u] = (_Float16)(v[u] * scale);
  *(uint4*)(Fh + ((size_t)b * S_ + j) * C_ + lane * 8) = o.q;
}

// m[b][k][j] = sigmoid(logits[b][k][idx[b][j]])
__global__ void gather_sigmoid_kernel(const float* __restrict__ logits,
                                      const int* __restrict__ indices,
                                      float* __restrict__ Msmp) {
  int t = blockIdx.x * blockDim.x + threadIdx.x;
  if (t >= B_ * K_ * S_) return;
  int b = t / (K_ * S_);
  int r = t % (K_ * S_);
  int k = r / S_;
  int j = r % S_;
  int p = indices[b * S_ + j];
  float x = logits[((size_t)b * K_ + k) * HW_ + p];
  Msmp[t] = 1.f / (1.f + __expf(-x));
}

// One wave per 16x16 tile of sim = F^T F, fused threshold + masked |dm| sum.
__global__ void __launch_bounds__(32)
affinity_tile_kernel(const _Float16* __restrict__ Fh,
                     const float* __restrict__ Msmp,
                     float* __restrict__ accum) {
  __shared__ float ms[K_ * TILE];   // m-values for the s-side of the tile

  int t0   = blockIdx.x * TILE;
  int s0   = blockIdx.y * TILE;
  int b    = blockIdx.z;
  int lane = threadIdx.x;
  int n    = lane & 15;             // column within tile (A.M / B.N / D.N)
  int h    = lane >> 4;             // lane half selects K sub-range

  // Stage m_sample[s-tile] into LDS (128 floats, 4 per lane).
#pragma unroll
  for (int q = 0; q < 4; ++q) {
    int idx = lane * 4 + q;
    int k  = idx >> 4;
    int sl = idx & 15;
    ms[idx] = Msmp[((size_t)b * K_ + k) * S_ + s0 + sl];
  }
  // Per-lane t-column m-values (t = t0 + n is fixed per lane in the D layout).
  float mt[K_];
#pragma unroll
  for (int k = 0; k < K_; ++k)
    mt[k] = Msmp[((size_t)b * K_ + k) * S_ + t0 + n];
  __syncthreads();

  // A fragment: 16x32 f16, lane holds row M = lane&15;
  //   VGPR j<4 : K = h*8 + 2j,+1 ; VGPR j>=4 : K = 16 + h*8 + 2(j-4),+1
  //   => two contiguous 8-half chunks at c0+h*8 and c0+16+h*8.
  // B fragment: 32x16 f16, lane holds column N = lane&15;
  //   same K mapping as A => two contiguous 8-half chunks at c0+h*8 and c0+16+h*8.
  const _Float16* Arow = Fh + ((size_t)b * S_ + s0 + n) * C_;
  const _Float16* Brow = Fh + ((size_t)b * S_ + t0 + n) * C_;

  v8f acc = {};
#pragma unroll
  for (int c0 = 0; c0 < C_; c0 += 32) {
    uint4 a0 = *(const uint4*)(Arow + c0 + h * 8);
    uint4 a1 = *(const uint4*)(Arow + c0 + 16 + h * 8);
    uint4 b0 = *(const uint4*)(Brow + c0 + h * 8);
    uint4 b1 = *(const uint4*)(Brow + c0 + 16 + h * 8);
    v8u au = {a0.x, a0.y, a0.z, a0.w, a1.x, a1.y, a1.z, a1.w};
    v8u bu = {b0.x, b0.y, b0.z, b0.w, b1.x, b1.y, b1.z, b1.w};
    v16h av = __builtin_bit_cast(v16h, au);
    v16h bv = __builtin_bit_cast(v16h, bu);
    acc = __builtin_amdgcn_wmma_f32_16x16x32_f16(
        /*neg_a=*/false, av, /*neg_b=*/false, bv,
        /*c_mod=*/(short)0, acc, /*reuse_a=*/false, /*reuse_b=*/false);
  }

  // D layout: lanes 0-15 -> M = r; lanes 16-31 -> M = r+8; N = lane&15.
  float numl = 0.f, denl = 0.f;
#pragma unroll
  for (int r = 0; r < 8; ++r) {
    int m = r + h * 8;
    float d = 0.f;
#pragma unroll
    for (int k = 0; k < K_; ++k) d += fabsf(ms[k * 16 + m] - mt[k]);
    if (acc[r] > SIM_TH) { denl += 1.f; numl += d; }
  }
#pragma unroll
  for (int off = 16; off >= 1; off >>= 1) {
    numl += __shfl_xor(numl, off, 32);
    denl += __shfl_xor(denl, off, 32);
  }
  if (lane == 0) {
    atomicAdd(&accum[0], numl);
    atomicAdd(&accum[1], denl);
  }
}

__global__ void finalize_kernel(const float* __restrict__ accum,
                                float* __restrict__ out) {
  out[0] = accum[0] / (accum[1] + 1e-6f);
}

extern "C" void kernel_launch(void* const* d_in, const int* in_sizes, int n_in,
                              void* d_out, int out_size, void* d_ws, size_t ws_size,
                              hipStream_t stream) {
  const float* logits = (const float*)d_in[0];   // (B,K,H,W) f32
  const float* feats  = (const float*)d_in[1];   // (B,C,H,W) f32
  const int*   idx    = (const int*)d_in[2];     // (B,S) i32

  _Float16* Fh   = (_Float16*)d_ws;
  float*    Msmp = (float*)((char*)d_ws + FH_BYTES);
  float*    acc  = (float*)((char*)d_ws + FH_BYTES + MS_BYTES);

  zero_accum_kernel<<<1, 32, 0, stream>>>(acc);
  gather_normalize_kernel<<<B_ * S_, 32, 0, stream>>>(feats, idx, Fh);
  gather_sigmoid_kernel<<<(B_ * K_ * S_ + 255) / 256, 256, 0, stream>>>(logits, idx, Msmp);

  dim3 grid(S_ / TILE, S_ / TILE, B_);
  affinity_tile_kernel<<<grid, 32, 0, stream>>>(Fh, Msmp, acc);

  finalize_kernel<<<1, 1, 0, stream>>>(acc, (float*)d_out);
}
